// MultiQueryAttention_63041529970960
// MI455X (gfx1250) — matrix operand order repaired and checked by reading, not compiled, not run on hardware
//
#include <hip/hip_runtime.h>
#include <hip/hip_bf16.h>

typedef __bf16 bf16_t;
typedef __attribute__((ext_vector_type(16))) __bf16 v16bf;
typedef __attribute__((ext_vector_type(8)))  __bf16 v8bf;
typedef __attribute__((ext_vector_type(8)))  float   v8f;

#define S_LEN   2048
#define D_MODEL 2048
#define HEADS   16
#define HDIM    128
#define BATCH   2

#define LDS_AS __attribute__((address_space(3)))

__device__ __forceinline__ v8bf ld8(const bf16_t* p) { return *(const v8bf*)p; }
__device__ __forceinline__ void st8(bf16_t* p, v8bf v) { *(v8bf*)p = v; }
__device__ __forceinline__ v16bf cat16(v8bf lo, v8bf hi) {
  v16bf r;
#pragma unroll
  for (int i = 0; i < 8; ++i) { r[i] = lo[i]; r[i + 8] = hi[i]; }
  return r;
}

// ---- CDNA5 async global->LDS DMA (ASYNCcnt-tracked), 16B per lane ---------
__device__ __forceinline__ unsigned lds_off32(void* p) {
  return (unsigned)(uintptr_t)(LDS_AS char*)p;
}
__device__ __forceinline__ void async_ld16(void* lds_dst, const void* gsrc) {
  asm volatile("global_load_async_to_lds_b128 %0, %1, off"
               :: "v"(lds_off32(lds_dst)),
                  "v"((unsigned long long)(uintptr_t)gsrc)
               : "memory");
}
__device__ __forceinline__ void wait_async0() {
  asm volatile("s_wait_asynccnt 0" ::: "memory");
}

// ---------------------------------------------------------------------------
// Weight transpose + fp32->bf16 convert:  W[K][N] -> Wt[N][K]
// ---------------------------------------------------------------------------
__global__ __launch_bounds__(256) void transpose_convert_w(
    const float* __restrict__ W, bf16_t* __restrict__ Wt, int K, int N) {
  __shared__ float tile[32][33];
  const int k0 = blockIdx.x * 32, n0 = blockIdx.y * 32;
  const int tx = threadIdx.x, ty = threadIdx.y;  // (32, 8)
#pragma unroll
  for (int i = ty; i < 32; i += 8)
    tile[i][tx] = W[(size_t)(k0 + i) * N + n0 + tx];
  __syncthreads();
#pragma unroll
  for (int i = ty; i < 32; i += 8)
    Wt[(size_t)(n0 + i) * K + k0 + tx] = (bf16_t)tile[tx][i];
}

// ---------------------------------------------------------------------------
// V transpose (bf16): Vp[B*S][HDIM] -> VpT[B][HDIM][S]
// ---------------------------------------------------------------------------
__global__ __launch_bounds__(256) void transpose_v(
    const bf16_t* __restrict__ Vp, bf16_t* __restrict__ VpT) {
  __shared__ bf16_t tile[32][33];
  const int b = blockIdx.z;
  const int s0 = blockIdx.x * 32, d0 = blockIdx.y * 32;
  const int tx = threadIdx.x, ty = threadIdx.y;  // (32, 8)
#pragma unroll
  for (int i = ty; i < 32; i += 8)
    tile[i][tx] = Vp[(size_t)(b * S_LEN + s0 + i) * HDIM + d0 + tx];
  __syncthreads();
#pragma unroll
  for (int i = ty; i < 32; i += 8)
    VpT[(size_t)(b * HDIM + d0 + i) * S_LEN + s0 + tx] = tile[tx][i];
}

// ---------------------------------------------------------------------------
// WMMA GEMM with bias:  C[M][N] = A[M][K] * Bt[N][K]^T + bias
// Double-buffered LDS, async global->LDS staging for bf16 operands,
// register-prefetch + convert for fp32 A. One barrier per K-step.
// Block tile 128x128, BK=32, 8 waves x (32x64) sub-tiles.
// ---------------------------------------------------------------------------
template <typename AT>
__global__ __launch_bounds__(256) void gemm_bias_wmma(
    const AT* __restrict__ A, const bf16_t* __restrict__ Bt,
    const float* __restrict__ bias, float* __restrict__ Cf,
    bf16_t* __restrict__ Cb, int M, int N, int K) {
  __shared__ __align__(16) bf16_t As[2][128 * 32];
  __shared__ __align__(16) bf16_t Bs[2][128 * 32];

  const int t = threadIdx.x;
  const int w = t >> 5, lane = t & 31;
  const int lm = lane & 15, lh = lane >> 4;
  const int m0 = blockIdx.y * 128, n0 = blockIdx.x * 128;
  const int wm = (w >> 1) * 32;  // wave row offset in block tile
  const int wn = (w & 1) * 64;   // wave col offset in block tile
  const int r = t >> 1;          // 0..127 staging row
  const int ch = (t & 1) * 16;   // 0/16 staging col half
  const int nk = K >> 5;

  v8f acc[2][4];
#pragma unroll
  for (int i = 0; i < 2; ++i)
#pragma unroll
    for (int j = 0; j < 4; ++j)
#pragma unroll
      for (int v = 0; v < 8; ++v) acc[i][j][v] = 0.f;

  // ---- prologue: stage tile 0 into buffer 0
  {
    const bf16_t* bsrc = Bt + (size_t)(n0 + r) * K + ch;
    async_ld16(&Bs[0][r * 32 + ch], bsrc);
    async_ld16(&Bs[0][r * 32 + ch + 8], bsrc + 8);
    if constexpr (sizeof(AT) == 2) {
      const bf16_t* asrc = (const bf16_t*)A + (size_t)(m0 + r) * K + ch;
      async_ld16(&As[0][r * 32 + ch], asrc);
      async_ld16(&As[0][r * 32 + ch + 8], asrc + 8);
    } else {
      const float* fsrc = (const float*)A + (size_t)(m0 + r) * K + ch;
      float f[16];
#pragma unroll
      for (int q = 0; q < 4; ++q) {
        float4 a = ((const float4*)fsrc)[q];
        f[q * 4 + 0] = a.x; f[q * 4 + 1] = a.y;
        f[q * 4 + 2] = a.z; f[q * 4 + 3] = a.w;
      }
      v8bf o0, o1;
#pragma unroll
      for (int q = 0; q < 8; ++q) { o0[q] = (bf16_t)f[q]; o1[q] = (bf16_t)f[q + 8]; }
      st8(&As[0][r * 32 + ch], o0);
      st8(&As[0][r * 32 + ch + 8], o1);
    }
  }
  wait_async0();
  __syncthreads();

  for (int k = 0; k < nk; ++k) {
    const int cur = k & 1, nxt = cur ^ 1;
    const bool has_next = (k + 1) < nk;

    // ---- issue next tile's global traffic before computing (overlap)
    float4 pre[4];
    if (has_next) {
      const int k1 = (k + 1) << 5;
      const bf16_t* bsrc = Bt + (size_t)(n0 + r) * K + k1 + ch;
      async_ld16(&Bs[nxt][r * 32 + ch], bsrc);
      async_ld16(&Bs[nxt][r * 32 + ch + 8], bsrc + 8);
      if constexpr (sizeof(AT) == 2) {
        const bf16_t* asrc = (const bf16_t*)A + (size_t)(m0 + r) * K + k1 + ch;
        async_ld16(&As[nxt][r * 32 + ch], asrc);
        async_ld16(&As[nxt][r * 32 + ch + 8], asrc + 8);
      } else {
        const float* fsrc = (const float*)A + (size_t)(m0 + r) * K + k1 + ch;
#pragma unroll
        for (int q = 0; q < 4; ++q) pre[q] = ((const float4*)fsrc)[q];
      }
    }

    // ---- fragments per ISA 16-bit layouts, from current buffer
    v16bf af[2], bfr[4];
#pragma unroll
    for (int i = 0; i < 2; ++i) {
      const int row = wm + i * 16 + lm;
      const int o = lh * 8;  // A: lanes<16 K{0..7,16..23}, lanes>=16 K{8..15,24..31}
      af[i] = cat16(ld8(&As[cur][row * 32 + o]), ld8(&As[cur][row * 32 + o + 16]));
    }
#pragma unroll
    for (int j = 0; j < 4; ++j) {
      const int n = wn + j * 16 + lm;
      const int o = lh * 8;  // B: lanes<16 K{0..7,16..23}, lanes>=16 K{8..15,24..31}
      bfr[j] = cat16(ld8(&Bs[cur][n * 32 + o]), ld8(&Bs[cur][n * 32 + o + 16]));
    }
#pragma unroll
    for (int i = 0; i < 2; ++i)
#pragma unroll
      for (int j = 0; j < 4; ++j)
        acc[i][j] = __builtin_amdgcn_wmma_f32_16x16x32_bf16(
            false, af[i], false, bfr[j], (short)0, acc[i][j], false, false);

    // ---- fp32 path: convert prefetched registers into next A buffer
    if constexpr (sizeof(AT) == 4) {
      if (has_next) {
        float f[16];
#pragma unroll
        for (int q = 0; q < 4; ++q) {
          f[q * 4 + 0] = pre[q].x; f[q * 4 + 1] = pre[q].y;
          f[q * 4 + 2] = pre[q].z; f[q * 4 + 3] = pre[q].w;
        }
        v8bf o0, o1;
#pragma unroll
        for (int q = 0; q < 8; ++q) { o0[q] = (bf16_t)f[q]; o1[q] = (bf16_t)f[q + 8]; }
        st8(&As[nxt][r * 32 + ch], o0);
        st8(&As[nxt][r * 32 + ch + 8], o1);
      }
    }
    wait_async0();
    __syncthreads();
  }

  // ---- epilogue: +bias, write fp32 and/or bf16
  float bval[4];
#pragma unroll
  for (int j = 0; j < 4; ++j) bval[j] = bias[n0 + wn + j * 16 + lm];
#pragma unroll
  for (int i = 0; i < 2; ++i)
#pragma unroll
    for (int j = 0; j < 4; ++j) {
      const int col = n0 + wn + j * 16 + lm;
#pragma unroll
      for (int v = 0; v < 8; ++v) {
        const int row = m0 + wm + i * 16 + lh * 8 + v;  // C layout: M = v (+8 hi half)
        const float val = acc[i][j][v] + bval[j];
        if (Cf) Cf[(size_t)row * N + col] = val;
        if (Cb) Cb[(size_t)row * N + col] = (bf16_t)val;
      }
    }
}

// ---------------------------------------------------------------------------
// Flash MQA: per (batch, head, 128-query tile). 8 waves x 16 query rows.
// Double-buffered K/V^T tiles staged with async global->LDS DMA.
// Qp[B*S][D] bf16, Kp[B*S][HDIM] bf16 (shared head), VpT[B][HDIM][S] bf16.
// ---------------------------------------------------------------------------
__global__ __launch_bounds__(256) void mqa_flash(
    const bf16_t* __restrict__ Qp, const bf16_t* __restrict__ Kp,
    const bf16_t* __restrict__ VpT, bf16_t* __restrict__ AO) {
  __shared__ __align__(16) bf16_t Kt[2][64 * HDIM];   // 2 x 16KB K tiles [sk][d]
  __shared__ __align__(16) bf16_t Vt[2][HDIM * 64];   // 2 x 16KB V^T tiles [d][sk]
  __shared__ __align__(16) bf16_t Pt[8][16 * 64];     // 16KB per-wave P tiles

  const int t = threadIdx.x, w = t >> 5, lane = t & 31;
  const int lm = lane & 15, lh = lane >> 4;
  const int b = blockIdx.z, h = blockIdx.y, q0 = blockIdx.x * 128;
  const float scale = 0.08838834764831845f;  // 1/sqrt(128)

  auto stage_kv = [&](int s0, int buf) {
    const bf16_t* kb8 = Kp + (size_t)(b * S_LEN + s0) * HDIM;  // contiguous tile
#pragma unroll
    for (int c = 0; c < 4; ++c) {
      const int e = (t + c * 256) * 8;
      async_ld16(&Kt[buf][e], kb8 + e);
    }
#pragma unroll
    for (int c = 0; c < 4; ++c) {
      const int e = (t + c * 256) * 8;
      const int d = e >> 6, col = e & 63;
      async_ld16(&Vt[buf][e], VpT + (size_t)(b * HDIM + d) * S_LEN + s0 + col);
    }
  };

  // preload Q fragments: 16 rows x 128 head-dims = 4 K-steps, kept in VGPRs
  const int qr = b * S_LEN + q0 + w * 16 + lm;
  const bf16_t* qrow = Qp + (size_t)qr * D_MODEL + h * HDIM;
  v16bf qf[4];
#pragma unroll
  for (int d0 = 0; d0 < 4; ++d0)
    qf[d0] = cat16(ld8(qrow + d0 * 32 + lh * 8), ld8(qrow + d0 * 32 + lh * 8 + 16));

  float mrow[8], lrow[8];
  v8f oacc[8];
#pragma unroll
  for (int v = 0; v < 8; ++v) { mrow[v] = -3.0e38f; lrow[v] = 0.f; }
#pragma unroll
  for (int i = 0; i < 8; ++i)
#pragma unroll
    for (int v = 0; v < 8; ++v) oacc[i][v] = 0.f;

  stage_kv(0, 0);
  wait_async0();
  __syncthreads();

  const int nj = S_LEN / 64;
  for (int j = 0; j < nj; ++j) {
    const int cur = j & 1;
    if (j + 1 < nj) stage_kv((j + 1) * 64, cur ^ 1);  // overlap with compute

    // ---- S = Q K^T : 16x64 per wave, contraction over Dh=128
    v8f sacc[4];
#pragma unroll
    for (int tn = 0; tn < 4; ++tn)
#pragma unroll
      for (int v = 0; v < 8; ++v) sacc[tn][v] = 0.f;
#pragma unroll
    for (int tn = 0; tn < 4; ++tn)
#pragma unroll
      for (int d0 = 0; d0 < 4; ++d0) {
        const int sk = tn * 16 + lm;
        v16bf kf = cat16(ld8(&Kt[cur][sk * HDIM + d0 * 32 + lh * 8]),
                         ld8(&Kt[cur][sk * HDIM + d0 * 32 + lh * 8 + 16]));
        sacc[tn] = __builtin_amdgcn_wmma_f32_16x16x32_bf16(
            false, qf[d0], false, kf, (short)0, sacc[tn], false, false);
      }

    // ---- online softmax (row stats per C-layout VGPR slot)
    float mnew[8], alpha[8], psum[8];
#pragma unroll
    for (int v = 0; v < 8; ++v) {
      float x = fmaxf(fmaxf(sacc[0][v], sacc[1][v]), fmaxf(sacc[2][v], sacc[3][v]));
      x *= scale;
#pragma unroll
      for (int off = 1; off < 16; off <<= 1) x = fmaxf(x, __shfl_xor(x, off, 32));
      mnew[v] = fmaxf(mrow[v], x);
      alpha[v] = __expf(mrow[v] - mnew[v]);
      psum[v] = 0.f;
    }
#pragma unroll
    for (int tn = 0; tn < 4; ++tn)
#pragma unroll
      for (int v = 0; v < 8; ++v) {
        const float p = __expf(sacc[tn][v] * scale - mnew[v]);
        psum[v] += p;
        Pt[w][(lh * 8 + v) * 64 + tn * 16 + lm] = (bf16_t)p;  // C-layout -> row-major
      }
#pragma unroll
    for (int v = 0; v < 8; ++v) {
      float s = psum[v];
#pragma unroll
      for (int off = 1; off < 16; off <<= 1) s += __shfl_xor(s, off, 32);
      lrow[v] = lrow[v] * alpha[v] + s;
      mrow[v] = mnew[v];
#pragma unroll
      for (int i = 0; i < 8; ++i) oacc[i][v] *= alpha[v];
    }

    // ---- O += P V : P(16x64) x V(64x128); same-wave DS ops are in-order
    v16bf pf[2];
#pragma unroll
    for (int ks = 0; ks < 2; ++ks)
      pf[ks] = cat16(ld8(&Pt[w][lm * 64 + ks * 32 + lh * 8]),
                     ld8(&Pt[w][lm * 64 + ks * 32 + lh * 8 + 16]));
#pragma unroll
    for (int t2 = 0; t2 < 8; ++t2)
#pragma unroll
      for (int ks = 0; ks < 2; ++ks) {
        const int dh = t2 * 16 + lm;
        v16bf vf = cat16(ld8(&Vt[cur][dh * 64 + ks * 32 + lh * 8]),
                         ld8(&Vt[cur][dh * 64 + ks * 32 + lh * 8 + 16]));
        oacc[t2] = __builtin_amdgcn_wmma_f32_16x16x32_bf16(
            false, pf[ks], false, vf, (short)0, oacc[t2], false, false);
      }

    wait_async0();   // my async DMAs into the other buffer are done
    __syncthreads(); // everyone done reading cur + loading nxt
  }

  // ---- epilogue: normalize, store bf16 into [B*S][H*HDIM]
#pragma unroll
  for (int t2 = 0; t2 < 8; ++t2)
#pragma unroll
    for (int v = 0; v < 8; ++v) {
      const int row = q0 + w * 16 + lh * 8 + v;
      const float o = oacc[t2][v] / lrow[v];
      AO[(size_t)(b * S_LEN + row) * D_MODEL + h * HDIM + t2 * 16 + lm] = (bf16_t)o;
    }
}

// ---------------------------------------------------------------------------
extern "C" void kernel_launch(void* const* d_in, const int* in_sizes, int n_in,
                              void* d_out, int out_size, void* d_ws,
                              size_t ws_size, hipStream_t stream) {
  const float* query = (const float*)d_in[0];
  const float* key   = (const float*)d_in[1];
  const float* value = (const float*)d_in[2];
  const float* wq = (const float*)d_in[3];
  const float* bq = (const float*)d_in[4];
  const float* wk = (const float*)d_in[5];
  const float* bk = (const float*)d_in[6];
  const float* wv = (const float*)d_in[7];
  const float* bv = (const float*)d_in[8];
  const float* wo = (const float*)d_in[9];
  const float* bo = (const float*)d_in[10];
  float* out = (float*)d_out;

  char* ws = (char*)d_ws;
  size_t off = 0;
  auto wsalloc = [&](size_t bytes) -> void* {
    void* p = ws + off;
    off = (off + bytes + 255) & ~(size_t)255;
    return p;
  };
  bf16_t* wqT = (bf16_t*)wsalloc((size_t)D_MODEL * D_MODEL * 2);
  bf16_t* wkT = (bf16_t*)wsalloc((size_t)HDIM * D_MODEL * 2);
  bf16_t* wvT = (bf16_t*)wsalloc((size_t)HDIM * D_MODEL * 2);
  bf16_t* woT = (bf16_t*)wsalloc((size_t)D_MODEL * D_MODEL * 2);
  bf16_t* Qp  = (bf16_t*)wsalloc((size_t)BATCH * S_LEN * D_MODEL * 2);
  bf16_t* Kp  = (bf16_t*)wsalloc((size_t)BATCH * S_LEN * HDIM * 2);
  bf16_t* Vp  = (bf16_t*)wsalloc((size_t)BATCH * S_LEN * HDIM * 2);
  bf16_t* VpT = (bf16_t*)wsalloc((size_t)BATCH * HDIM * S_LEN * 2);
  bf16_t* AO  = (bf16_t*)wsalloc((size_t)BATCH * S_LEN * D_MODEL * 2);

  const int M = BATCH * S_LEN;  // 4096 token rows
  dim3 tb(32, 8);

  transpose_convert_w<<<dim3(D_MODEL / 32, D_MODEL / 32), tb, 0, stream>>>(wq, wqT, D_MODEL, D_MODEL);
  transpose_convert_w<<<dim3(D_MODEL / 32, HDIM / 32), tb, 0, stream>>>(wk, wkT, D_MODEL, HDIM);
  transpose_convert_w<<<dim3(D_MODEL / 32, HDIM / 32), tb, 0, stream>>>(wv, wvT, D_MODEL, HDIM);
  transpose_convert_w<<<dim3(D_MODEL / 32, D_MODEL / 32), tb, 0, stream>>>(wo, woT, D_MODEL, D_MODEL);

  gemm_bias_wmma<float><<<dim3(D_MODEL / 128, M / 128), 256, 0, stream>>>(
      query, wqT, bq, nullptr, Qp, M, D_MODEL, D_MODEL);
  gemm_bias_wmma<float><<<dim3(1, M / 128), 256, 0, stream>>>(
      key, wkT, bk, nullptr, Kp, M, HDIM, D_MODEL);
  gemm_bias_wmma<float><<<dim3(1, M / 128), 256, 0, stream>>>(
      value, wvT, bv, nullptr, Vp, M, HDIM, D_MODEL);

  transpose_v<<<dim3(S_LEN / 32, HDIM / 32, BATCH), tb, 0, stream>>>(Vp, VpT);

  mqa_flash<<<dim3(S_LEN / 128, HEADS, BATCH), 256, 0, stream>>>(Qp, Kp, VpT, AO);

  gemm_bias_wmma<bf16_t><<<dim3(D_MODEL / 128, M / 128), 256, 0, stream>>>(
      AO, woT, bo, out, nullptr, M, D_MODEL, D_MODEL);
}
